// HierarchicalAttention_7421703488080
// MI455X (gfx1250) — compile-verified, matrix operand order checked by reading, not run on hardware
//
#include <hip/hip_runtime.h>
#include <hip/hip_bf16.h>
#include <math.h>

typedef __attribute__((ext_vector_type(16))) _Float16 v16h;
typedef __attribute__((ext_vector_type(8)))  float    v8f;

// ---------------------------------------------------------------------------
// WMMA helper: D = A(16x32 f16) * B(32x16 f16) + C (f32)
// ---------------------------------------------------------------------------
__device__ __forceinline__ v8f wmma_f16(v16h a, v16h b, v8f c) {
  return __builtin_amdgcn_wmma_f32_16x16x32_f16(
      false, a, false, b, (short)0, c, false, false);
}

// ---------------------------------------------------------------------------
// Async global->LDS copy (CDNA5 GLOBAL_LOAD_ASYNC_TO_LDS_B128, ASYNCcnt).
// 16 bytes per lane; EXEC-masked. LDS destination is a 32-bit LDS byte offset.
// ---------------------------------------------------------------------------
__device__ __forceinline__ unsigned lds_addr(const void* p) {
  return (unsigned)(uintptr_t)(__attribute__((address_space(3))) const void*)p;
}
__device__ __forceinline__ void async_copy_b128(unsigned lds_dst, const void* gsrc) {
  asm volatile("global_load_async_to_lds_b128 %0, %1, off"
               :: "v"(lds_dst), "v"(gsrc) : "memory");
}
__device__ __forceinline__ void wait_async0() {
#if __has_builtin(__builtin_amdgcn_s_wait_asynccnt)
  __builtin_amdgcn_s_wait_asynccnt(0);
#else
  asm volatile("s_wait_asynccnt 0x0" ::: "memory");
#endif
}

// Fragment loader from a [major][k] f16 buffer (k contiguous, leading dim ld
// in halfs; ld*2 multiple of 16, base 16B-aligned -> two ds_load_b128 / lane).
// ISA A layout: lanes 0-15 hold K {0..7,16..23}; lanes 16-31 hold {8..15,24..31}.
__device__ __forceinline__ v16h load_frag(const _Float16* P, int ld, int idx0,
                                          int k0, int lane) {
  const _Float16* p = P + (idx0 + (lane & 15)) * ld + k0 + ((lane & 16) ? 8 : 0);
  v16h a;
#pragma unroll
  for (int v = 0; v < 8; ++v) {
    int kb = ((v & 3) << 1) + ((v >> 2) << 4);  // 0,2,4,6,16,18,20,22
    a[2 * v]     = p[kb];
    a[2 * v + 1] = p[kb + 1];
  }
  return a;
}

// ---------------------------------------------------------------------------
// One-time conversions: f32 -> f16 (flat) and f32 [K][N] -> f16 W^T [N][K]
// ---------------------------------------------------------------------------
__global__ __launch_bounds__(256) void cvt_f16(const float* __restrict__ in,
                                               _Float16* __restrict__ out,
                                               size_t n) {
  size_t i = ((size_t)blockIdx.x * 256 + threadIdx.x) * 8;
  if (i + 8 <= n) {
#pragma unroll
    for (int j = 0; j < 8; ++j) out[i + j] = (_Float16)in[i + j];
  }
}

__global__ __launch_bounds__(256) void transpose_cvt(const float* __restrict__ W,
                                                     _Float16* __restrict__ Wt,
                                                     int K, int N) {
  __shared__ __align__(16) _Float16 t[64 * 72];
  const int tid = threadIdx.x;
  const int k0 = blockIdx.y * 64, n0 = blockIdx.x * 64;
  for (int i = tid; i < 512; i += 256) {          // 64k x 64n / 8
    int k = i >> 3, nb = (i & 7) * 8;
    const float* src = &W[(size_t)(k0 + k) * N + n0 + nb];
#pragma unroll
    for (int j = 0; j < 8; ++j) t[(nb + j) * 72 + k] = (_Float16)src[j];
  }
  __syncthreads();
  for (int i = tid; i < 512; i += 256) {
    int n = i >> 3, kb = (i & 7) * 8;
    _Float16* dst = &Wt[(size_t)(n0 + n) * K + k0 + kb];
#pragma unroll
    for (int j = 0; j < 8; ++j) dst[j] = t[n * 72 + kb + j];
  }
}

// ---------------------------------------------------------------------------
// GEMM: Y[M,N] = (X (+X2)) [M,K]f16 * W^T[N,K]f16 + bias(f32)
// 64(M) x 128(N) block tile, 8 waves (4xM, 2xN), 4 WMMA / wave / K-step.
// Double-buffered LDS filled with async global->LDS DMA (ASYNCcnt).
// ---------------------------------------------------------------------------
template <bool HAS2, bool OUTF16>
__global__ __launch_bounds__(256) void gemm_f16(
    const _Float16* __restrict__ X, const _Float16* __restrict__ X2,
    const _Float16* __restrict__ Wt, const float* __restrict__ bias,
    float* __restrict__ Yf, _Float16* __restrict__ Yh, int M, int K, int N) {
  __shared__ __align__(16) _Float16 As[2][64 * 40];    // [m][k]
  __shared__ __align__(16) _Float16 Bs[2][128 * 40];   // [n][k]

  const int tid = threadIdx.x, lane = tid & 31, wave = tid >> 5;
  const int m0 = blockIdx.y * 64, n0 = blockIdx.x * 128;
  const int msub = (wave & 3) * 16, nsub = (wave >> 2) * 64;

  v8f c[4] = {{}, {}, {}, {}};

  const int ar = tid >> 2, ak = (tid & 3) * 8;       // A: 64x32, 16B/thread
  const int bn = tid >> 2, bk = (tid & 3) * 8;       // B: 128x32, 2x16B/thread
  const unsigned aDst[2] = {lds_addr(&As[0][ar * 40 + ak]),
                            lds_addr(&As[1][ar * 40 + ak])};
  const unsigned bDst0[2] = {lds_addr(&Bs[0][bn * 40 + bk]),
                             lds_addr(&Bs[1][bn * 40 + bk])};
  const unsigned bDst1[2] = {lds_addr(&Bs[0][(bn + 64) * 40 + bk]),
                             lds_addr(&Bs[1][(bn + 64) * 40 + bk])};

  auto issue = [&](int kt, int buf) {
    if constexpr (HAS2) {
      const _Float16* s1 = &X[(size_t)(m0 + ar) * K + kt + ak];
      const _Float16* s2 = &X2[(size_t)(m0 + ar) * K + kt + ak];
      _Float16* dst = &As[buf][ar * 40 + ak];
#pragma unroll
      for (int j = 0; j < 8; ++j) dst[j] = s1[j] + s2[j];
    } else {
      async_copy_b128(aDst[buf], &X[(size_t)(m0 + ar) * K + kt + ak]);
    }
    async_copy_b128(bDst0[buf], &Wt[(size_t)(n0 + bn) * K + kt + bk]);
    async_copy_b128(bDst1[buf], &Wt[(size_t)(n0 + bn + 64) * K + kt + bk]);
  };

  issue(0, 0);
  wait_async0();
  __syncthreads();
  for (int kt = 0; kt < K; kt += 32) {
    const int cur = (kt >> 5) & 1;
    if (kt + 32 < K) issue(kt + 32, cur ^ 1);
    v16h a = load_frag(&As[cur][0], 40, msub, 0, lane);
#pragma unroll
    for (int j = 0; j < 4; ++j) {
      v16h b = load_frag(&Bs[cur][0], 40, nsub + j * 16, 0, lane);
      c[j] = wmma_f16(a, b, c[j]);
    }
    wait_async0();
    __syncthreads();
  }

  float bb[4];
#pragma unroll
  for (int j = 0; j < 4; ++j) bb[j] = bias[n0 + nsub + j * 16 + (lane & 15)];
#pragma unroll
  for (int v = 0; v < 8; ++v) {
    int r = m0 + msub + v + ((lane & 16) ? 8 : 0);
#pragma unroll
    for (int j = 0; j < 4; ++j) {
      int cN = n0 + nsub + j * 16 + (lane & 15);
      float val = c[j][v] + bb[j];
      if (OUTF16) Yh[(size_t)r * N + cN] = (_Float16)val;
      else        Yf[(size_t)r * N + cN] = val;
    }
  }
}

// ---------------------------------------------------------------------------
// Sentence-anchor attention: out = softmax(Q Ks^T * s) Vs  (A = 64 anchors)
// f16 in / f16 out. Block = 4 waves x 16 queries.
// ---------------------------------------------------------------------------
__global__ __launch_bounds__(128) void sent_attn(
    const _Float16* __restrict__ Q, const _Float16* __restrict__ Ksg,
    const _Float16* __restrict__ Vsg, _Float16* __restrict__ Out, int L, int H) {
  __shared__ __align__(16) _Float16 Kt[64 * 72];     // [anchor][d]
  __shared__ __align__(16) _Float16 Vt[64 * 72];     // [d][anchor]
  __shared__ __align__(16) _Float16 Pr[4][16 * 72];  // per-wave [row][anchor]

  const int tid = threadIdx.x, lane = tid & 31, wave = tid >> 5;
  const int bh = blockIdx.y, h = bh % H, b = bh / H;
  const int D = H * 64;
  const int q0 = blockIdx.x * 64 + wave * 16;
  const float scale = 0.125f;
  const int rrow = (lane & 16) ? 8 : 0;

  for (int i = tid; i < 512; i += 128) {             // 64a x 64d / 8
    int a = i >> 3, db = (i & 7) * 8;
    async_copy_b128(lds_addr(&Kt[a * 72 + db]), &Ksg[(size_t)a * D + h * 64 + db]);
    const _Float16* vs = &Vsg[(size_t)a * D + h * 64 + db];
#pragma unroll
    for (int j = 0; j < 8; ++j) Vt[(db + j) * 72 + a] = vs[j];
  }
  wait_async0();
  __syncthreads();

  const _Float16* qb = Q + ((size_t)b * L + q0) * D + h * 64;
  v16h aq0 = load_frag(qb, D, 0, 0, lane);
  v16h aq1 = load_frag(qb, D, 0, 32, lane);

  v8f s[4] = {{}, {}, {}, {}};
#pragma unroll
  for (int j = 0; j < 4; ++j) {
    s[j] = wmma_f16(aq0, load_frag(Kt, 72, j * 16, 0, lane), s[j]);
    s[j] = wmma_f16(aq1, load_frag(Kt, 72, j * 16, 32, lane), s[j]);
  }

#pragma unroll
  for (int v = 0; v < 8; ++v) {
    float rmax = -1e30f;
#pragma unroll
    for (int j = 0; j < 4; ++j) {
      float val = s[j][v] * scale;
      s[j][v] = val;
      rmax = fmaxf(rmax, val);
    }
#pragma unroll
    for (int off = 1; off < 16; off <<= 1)
      rmax = fmaxf(rmax, __shfl_xor(rmax, off, 32));
    float rsum = 0.f;
#pragma unroll
    for (int j = 0; j < 4; ++j) {
      float p = __expf(s[j][v] - rmax);
      s[j][v] = p;
      rsum += p;
    }
#pragma unroll
    for (int off = 1; off < 16; off <<= 1) rsum += __shfl_xor(rsum, off, 32);
    float inv = 1.0f / rsum;
#pragma unroll
    for (int j = 0; j < 4; ++j)
      Pr[wave][(v + rrow) * 72 + j * 16 + (lane & 15)] = (_Float16)(s[j][v] * inv);
  }

  v16h ap0 = load_frag(&Pr[wave][0], 72, 0, 0, lane);
  v16h ap1 = load_frag(&Pr[wave][0], 72, 0, 32, lane);
  v8f o[4] = {{}, {}, {}, {}};
#pragma unroll
  for (int j = 0; j < 4; ++j) {
    o[j] = wmma_f16(ap0, load_frag(Vt, 72, j * 16, 0, lane), o[j]);
    o[j] = wmma_f16(ap1, load_frag(Vt, 72, j * 16, 32, lane), o[j]);
  }

#pragma unroll
  for (int v = 0; v < 8; ++v) {
    int r = q0 + v + rrow;
#pragma unroll
    for (int j = 0; j < 4; ++j) {
      int d = j * 16 + (lane & 15);
      Out[((size_t)b * L + r) * D + h * 64 + d] = (_Float16)o[j][v];
    }
  }
}

// ---------------------------------------------------------------------------
// Banded local attention, flash-style online softmax over 64-key tiles.
// Band |q-k| <= 128 plus key attention_mask. f16 in / f16 out.
// ---------------------------------------------------------------------------
__global__ __launch_bounds__(128) void local_attn(
    const _Float16* __restrict__ Q, const _Float16* __restrict__ Kh,
    const _Float16* __restrict__ Vh, const int* __restrict__ amask,
    _Float16* __restrict__ Out, int L, int H) {
  __shared__ __align__(16) _Float16 Kt[64 * 72];     // [key][d]
  __shared__ __align__(16) _Float16 Vt[64 * 72];     // [d][key]
  __shared__ __align__(16) _Float16 Pr[4][16 * 72];  // per-wave [row][key]

  const int tid = threadIdx.x, lane = tid & 31, wave = tid >> 5;
  const int bh = blockIdx.y, h = bh % H, b = bh / H;
  const int D = H * 64;
  const int q0b = blockIdx.x * 64;
  const int q0w = q0b + wave * 16;
  const float scale = 0.125f;
  const int rrow = (lane & 16) ? 8 : 0;

  const _Float16* qb = Q + ((size_t)b * L + q0w) * D + h * 64;
  v16h aq0 = load_frag(qb, D, 0, 0, lane);
  v16h aq1 = load_frag(qb, D, 0, 32, lane);

  float m[8], lsum[8];
#pragma unroll
  for (int v = 0; v < 8; ++v) { m[v] = -1e30f; lsum[v] = 0.f; }
  v8f o[4] = {{}, {}, {}, {}};

  int kt0 = q0b - 128; if (kt0 < 0) kt0 = 0;        // multiple of 64
  int kte = q0b + 64 + 128; if (kte > L) kte = L;   // exclusive

  for (int kt = kt0; kt < kte; kt += 64) {
    for (int i = tid; i < 512; i += 128) {          // 64key x 64d / 8
      int key = i >> 3, db = (i & 7) * 8;
      int gk = kt + key;
      if (gk < L) {
        async_copy_b128(lds_addr(&Kt[key * 72 + db]),
                        &Kh[((size_t)b * L + gk) * D + h * 64 + db]);
        const _Float16* vs = &Vh[((size_t)b * L + gk) * D + h * 64 + db];
#pragma unroll
        for (int j = 0; j < 8; ++j) Vt[(db + j) * 72 + key] = vs[j];
      } else {
        _Float16* kd = &Kt[key * 72 + db];
#pragma unroll
        for (int j = 0; j < 8; ++j) {
          kd[j] = (_Float16)0.f;
          Vt[(db + j) * 72 + key] = (_Float16)0.f;
        }
      }
    }
    wait_async0();
    __syncthreads();

    v8f s[4] = {{}, {}, {}, {}};
#pragma unroll
    for (int j = 0; j < 4; ++j) {
      s[j] = wmma_f16(aq0, load_frag(Kt, 72, j * 16, 0, lane), s[j]);
      s[j] = wmma_f16(aq1, load_frag(Kt, 72, j * 16, 32, lane), s[j]);
    }

    int kcol[4]; bool kok[4];
#pragma unroll
    for (int j = 0; j < 4; ++j) {
      kcol[j] = kt + j * 16 + (lane & 15);
      bool inr = kcol[j] < L;
      kok[j] = inr && (amask[b * L + (inr ? kcol[j] : 0)] != 0);
    }

#pragma unroll
    for (int v = 0; v < 8; ++v) {
      int q = q0w + v + rrow;
      float rmax = -1e30f;
#pragma unroll
      for (int j = 0; j < 4; ++j) {
        int dq = q - kcol[j]; if (dq < 0) dq = -dq;
        float val = (kok[j] && dq <= 128) ? s[j][v] * scale : -10000.0f;
        s[j][v] = val;
        rmax = fmaxf(rmax, val);
      }
#pragma unroll
      for (int off = 1; off < 16; off <<= 1)
        rmax = fmaxf(rmax, __shfl_xor(rmax, off, 32));
      float mn = fmaxf(m[v], rmax);
      float f = __expf(m[v] - mn);
      m[v] = mn;
      float rsum = 0.f;
#pragma unroll
      for (int j = 0; j < 4; ++j) {
        float p = __expf(s[j][v] - mn);
        s[j][v] = p;
        rsum += p;
      }
#pragma unroll
      for (int off = 1; off < 16; off <<= 1) rsum += __shfl_xor(rsum, off, 32);
      lsum[v] = lsum[v] * f + rsum;
#pragma unroll
      for (int j = 0; j < 4; ++j) {
        o[j][v] *= f;
        Pr[wave][(v + rrow) * 72 + j * 16 + (lane & 15)] = (_Float16)s[j][v];
      }
    }

    v16h ap0 = load_frag(&Pr[wave][0], 72, 0, 0, lane);
    v16h ap1 = load_frag(&Pr[wave][0], 72, 0, 32, lane);
#pragma unroll
    for (int j = 0; j < 4; ++j) {
      o[j] = wmma_f16(ap0, load_frag(Vt, 72, j * 16, 0, lane), o[j]);
      o[j] = wmma_f16(ap1, load_frag(Vt, 72, j * 16, 32, lane), o[j]);
    }
    __syncthreads();
  }

#pragma unroll
  for (int v = 0; v < 8; ++v) {
    int r = q0w + v + rrow;
    float inv = 1.0f / lsum[v];
#pragma unroll
    for (int j = 0; j < 4; ++j) {
      int d = j * 16 + (lane & 15);
      Out[((size_t)b * L + r) * D + h * 64 + d] = (_Float16)(o[j][v] * inv);
    }
  }
}

// ---------------------------------------------------------------------------
// Launcher
// ---------------------------------------------------------------------------
extern "C" void kernel_launch(void* const* d_in, const int* in_sizes, int n_in,
                              void* d_out, int out_size, void* d_ws, size_t ws_size,
                              hipStream_t stream) {
  (void)in_sizes; (void)n_in; (void)out_size; (void)ws_size;
  const float* x           = (const float*)d_in[0];
  const int*   amask       = (const int*)d_in[1];
  const float* sent_anchor = (const float*)d_in[2];
  const float* Wq_tok  = (const float*)d_in[4];  const float* bq_tok  = (const float*)d_in[5];
  const float* Wk_sent = (const float*)d_in[6];  const float* bk_sent = (const float*)d_in[7];
  const float* Wv_sent = (const float*)d_in[8];  const float* bv_sent = (const float*)d_in[9];
  const float* Wq_loc  = (const float*)d_in[10]; const float* bq_loc  = (const float*)d_in[11];
  const float* Wk_loc  = (const float*)d_in[12]; const float* bk_loc  = (const float*)d_in[13];
  const float* Wv_loc  = (const float*)d_in[14]; const float* bv_loc  = (const float*)d_in[15];
  const float* Wout    = (const float*)d_in[16]; const float* bout    = (const float*)d_in[17];
  float* out = (float*)d_out;

  const int Bz = 2, L = 2048, Dm = 1024, H = 16;
  const int M = Bz * L;                // 4096
  const size_t big = (size_t)M * Dm;   // 4,194,304
  const size_t ww  = (size_t)Dm * Dm;  // 1,048,576

  _Float16* hws = (_Float16*)d_ws;
  size_t off = 0;
  auto halloc = [&](size_t n) { _Float16* p = hws + off; off += n; return p; };
  _Float16* xh      = halloc(big);
  _Float16* sah     = halloc((size_t)64 * Dm);
  _Float16* WtQtok  = halloc(ww);
  _Float16* WtKsent = halloc(ww);
  _Float16* WtVsent = halloc(ww);
  _Float16* WtQloc  = halloc(ww);
  _Float16* WtKloc  = halloc(ww);
  _Float16* WtVloc  = halloc(ww);
  _Float16* WtOut   = halloc(ww);
  _Float16* q_tok   = halloc(big);
  _Float16* q_l     = halloc(big);
  _Float16* k_l     = halloc(big);
  _Float16* v_l     = halloc(big);
  _Float16* k_sent  = halloc((size_t)64 * Dm);
  _Float16* v_sent  = halloc((size_t)64 * Dm);
  _Float16* ctx     = halloc(big);
  _Float16* lctx    = halloc(big);

  // one-time f16 conversions / weight transposes
  cvt_f16<<<dim3((unsigned)(big / 2048)), dim3(256), 0, stream>>>(x, xh, big);
  cvt_f16<<<dim3(32), dim3(256), 0, stream>>>(sent_anchor, sah, (size_t)64 * Dm);
  dim3 grdT(Dm / 64, Dm / 64);
  transpose_cvt<<<grdT, 256, 0, stream>>>(Wq_tok,  WtQtok,  Dm, Dm);
  transpose_cvt<<<grdT, 256, 0, stream>>>(Wk_sent, WtKsent, Dm, Dm);
  transpose_cvt<<<grdT, 256, 0, stream>>>(Wv_sent, WtVsent, Dm, Dm);
  transpose_cvt<<<grdT, 256, 0, stream>>>(Wq_loc,  WtQloc,  Dm, Dm);
  transpose_cvt<<<grdT, 256, 0, stream>>>(Wk_loc,  WtKloc,  Dm, Dm);
  transpose_cvt<<<grdT, 256, 0, stream>>>(Wv_loc,  WtVloc,  Dm, Dm);
  transpose_cvt<<<grdT, 256, 0, stream>>>(Wout,    WtOut,   Dm, Dm);

  dim3 grdP(Dm / 128, M / 64);  // projections
  dim3 grdS(Dm / 128, 1);       // anchor projections (M = 64)
  const _Float16* nul = nullptr;
  gemm_f16<false, true><<<grdP, 256, 0, stream>>>(xh, nul, WtQtok, bq_tok, nullptr, q_tok, M, Dm, Dm);
  gemm_f16<false, true><<<grdS, 256, 0, stream>>>(sah, nul, WtKsent, bk_sent, nullptr, k_sent, 64, Dm, Dm);
  gemm_f16<false, true><<<grdS, 256, 0, stream>>>(sah, nul, WtVsent, bv_sent, nullptr, v_sent, 64, Dm, Dm);
  gemm_f16<false, true><<<grdP, 256, 0, stream>>>(xh, nul, WtQloc, bq_loc, nullptr, q_l, M, Dm, Dm);
  gemm_f16<false, true><<<grdP, 256, 0, stream>>>(xh, nul, WtKloc, bk_loc, nullptr, k_l, M, Dm, Dm);
  gemm_f16<false, true><<<grdP, 256, 0, stream>>>(xh, nul, WtVloc, bv_loc, nullptr, v_l, M, Dm, Dm);

  dim3 grdA(L / 64, Bz * H);
  sent_attn<<<grdA, 128, 0, stream>>>(q_tok, k_sent, v_sent, ctx, L, H);
  local_attn<<<grdA, 128, 0, stream>>>(q_l, k_l, v_l, amask, lctx, L, H);

  // out = (local_ctx + ctx) @ Wout + bout  (fused f16 add via HAS2)
  gemm_f16<true, false><<<grdP, 256, 0, stream>>>(lctx, ctx, WtOut, bout, out, nullptr, M, Dm, Dm);
}
